// CMPNEncoder_22368189678083
// MI455X (gfx1250) — matrix operand order repaired and checked by reading, not compiled, not run on hardware
//
#include <hip/hip_runtime.h>

typedef __attribute__((ext_vector_type(16))) _Float16 v16h;
typedef __attribute__((ext_vector_type(8)))  float    v8f;

#define N_ATOMS 400000
#define KDIM    133
#define NDIM    300
#define KPAD    160   // 5 * 32
#define BM      64    // rows per block (each wave: 2 subtiles of 16 rows)
#define NFG     80000
#define MAXFG   8
#define NFUNC   100
#define HSUB    100   // hidden-column chunk per gather block

// W fragment buffer: [ks 0..4][tile 0..19][lane 0..31][vv 0..7] packed f16 pairs
#define WFRAG_UINTS (5 * 20 * 32 * 8)   // 25600 uints = 100 KB (L2-resident)

static __device__ __forceinline__ unsigned short f2h_bits(float f) {
  _Float16 h = (_Float16)f;                 // RNE convert
  union { _Float16 h; unsigned short s; } u;
  u.h = h;
  return u.s;
}

// ---------------------------------------------------------------------------
// Kernel W-prep (one-time, 100 blocks): convert W [133x300] f32 into a contiguous-k
// v_wmma_f32_16x16x32_f16 B-fragment layout so GEMM blocks can load fragments
// with coalesced global_load_b128 (no per-block conversion, no LDS staging).
// Entry ((ks*20 + t)*32 + lane)*8 + vv = pack(W[K0+kk][n], W[K0+kk+1][n]),
// kk = (lane>>4)*16 + 2*vv, n = t*16 + (lane&15), K0 = ks*32.
// ---------------------------------------------------------------------------
__global__ __launch_bounds__(256) void wprep(const float* __restrict__ W,
                                             unsigned int* __restrict__ wfrag) {
  const int i = blockIdx.x * 256 + threadIdx.x;
  if (i >= WFRAG_UINTS) return;
  const int vv = i & 7;
  const int ln = (i >> 3) & 31;
  const int t  = (i >> 8) % 20;
  const int ks = i / 5120;
  const int kk = (ln >> 4) * 16 + 2 * vv;
  const int n  = t * 16 + (ln & 15);
  const int kg = ks * 32 + kk;
  float lo = 0.0f, hi = 0.0f;
  if (n < NDIM) {
    if (kg < KDIM)     lo = W[(size_t)kg * NDIM + n];
    if (kg + 1 < KDIM) hi = W[(size_t)(kg + 1) * NDIM + n];
  }
  wfrag[i] = (unsigned int)f2h_bits(lo) | ((unsigned int)f2h_bits(hi) << 16);
}

// ---------------------------------------------------------------------------
// Kernel 1: input_atom = f_atoms @ W  via v_wmma_f32_16x16x32_f16
// Block: 256 threads = 8 waves, tile 64(M) x 320(N), K padded to 160.
// Wave (wm, wn): wm picks a 32-row half (2 A-subtiles), wn picks 80 cols
// (5 B-tiles); each B fragment feeds 2 WMMAs. A is staged once to LDS
// (f32->f16, coalesced); B fragments come straight from the L2-resident
// pre-swizzled buffer as global_load_b128 pairs. No barriers in the k-loop.
// ---------------------------------------------------------------------------
__global__ __launch_bounds__(256) void gemm_f16_wmma(
    const float* __restrict__ A,              // [400000 x 133]
    const unsigned int* __restrict__ wfrag,   // [WFRAG_UINTS]
    float* __restrict__ out) {                // [400000 x 300]
  __shared__ unsigned short sA[BM * KPAD];    // [row][k] f16 bits, 20 KB

  const int tid  = threadIdx.x;
  const int row0 = blockIdx.x * BM;           // 400000/64 = 6250 blocks, exact

  // Stage the whole 64x160 A block once (f32 -> f16), coalesced along k.
  for (int i = tid; i < BM * KPAD; i += 256) {
    const int r = i / KPAD, k = i % KPAD;
    const float v = (k < KDIM) ? A[(size_t)(row0 + r) * KDIM + k] : 0.0f;
    sA[i] = f2h_bits(v);
  }
  __syncthreads();

  const int wid   = tid >> 5;
  const int lane  = tid & 31;
  const int lhalf = lane >> 4;   // 0: lanes 0-15, 1: lanes 16-31
  const int lm    = lane & 15;
  const int wm    = wid >> 2;          // 0..1 -> 32-row half
  const int wn    = wid & 3;           // 0..3 -> 80-col strip
  const int wrow0 = wm * 32;
  const int wcol0 = wn * 80;

  v8f acc[2][5] = {};

  for (int ks = 0; ks < 5; ++ks) {
    const int K0 = ks * 32;
    if (ks < 4 && tid < 80) {  // pull next fragment chunk toward L2/L0
      __builtin_prefetch(&wfrag[(ks + 1) * 5120 + tid * 64], 0, 1);
    }

    // A fragments (contiguous k order, matching wfrag): per VGPR vv the K pair
    // is 2vv, +16 for lanes 16-31. Consecutive pairs across the 8 VGPRs
    // -> merges to ds_load_b128 x2.
    union { v16h v; unsigned int u[8]; } af[2];
    #pragma unroll
    for (int s = 0; s < 2; ++s) {
      const unsigned int* ar32 =
          (const unsigned int*)&sA[(wrow0 + s * 16 + lm) * KPAD];
      #pragma unroll
      for (int vv = 0; vv < 8; ++vv) {
        const int kb = lhalf * 16 + 2 * vv;
        af[s].u[vv] = ar32[(K0 + kb) >> 1];
      }
    }

    #pragma unroll
    for (int t = 0; t < 5; ++t) {
      // B fragment: 1 KB contiguous per wave -> 2x global_load_b128 (L2 hit).
      union { v16h v; uint4 q[2]; } bf;
      const uint4* bp = (const uint4*)
          &wfrag[(((ks * 20 + wn * 5 + t) * 32) + lane) * 8];
      bf.q[0] = bp[0];
      bf.q[1] = bp[1];
      acc[0][t] = __builtin_amdgcn_wmma_f32_16x16x32_f16(
          false, af[0].v, false, bf.v, (short)0, acc[0][t], false, false);
      acc[1][t] = __builtin_amdgcn_wmma_f32_16x16x32_f16(
          false, af[1].v, false, bf.v, (short)0, acc[1][t], false, false);
    }
  }

  // C/D layout: VGPR e -> M = e + lhalf*8, N = lane column.
  #pragma unroll
  for (int t = 0; t < 5; ++t) {
    const int colb = wcol0 + t * 16 + lm;
    if (colb < NDIM) {
      #pragma unroll
      for (int s = 0; s < 2; ++s) {
        #pragma unroll
        for (int e = 0; e < 8; ++e) {
          const int r = row0 + wrow0 + s * 16 + lhalf * 8 + e;
          out[(size_t)r * NDIM + colb] = acc[s][t][e];
        }
      }
    }
  }
}

// ---------------------------------------------------------------------------
// Kernel 2: fg_emb = padded[func2atom].sum(1); out[mapping[fg]] += fg_emb
// grid.y selects a 100-wide hidden chunk; per-block LDS accumulator 100x100.
// ---------------------------------------------------------------------------
__global__ __launch_bounds__(256) void gather_segsum(
    const float* __restrict__ inAtom,     // [400000 x 300]
    const int*   __restrict__ f2a,        // [80000 x 8], 0 == padding
    const int*   __restrict__ mapping,    // [80000]
    float*       __restrict__ outp) {     // [100 x 300]
  __shared__ float acc[NFUNC * HSUB];     // 40 KB
  __shared__ int   sIdx[2][MAXFG];
  __shared__ int   sMap[2];

  const int tid = threadIdx.x;
  for (int i = tid; i < NFUNC * HSUB; i += 256) acc[i] = 0.0f;

  const int colBase = blockIdx.y * HSUB;  // 0,100,200
  const int sub = tid >> 7;               // two fg slots per iteration
  const int c   = tid & 127;
  __syncthreads();

  for (int fg0 = blockIdx.x * 2; fg0 < NFG; fg0 += gridDim.x * 2) {
    const int fg = fg0 + sub;
    if (fg < NFG) {
      if (c < MAXFG)  sIdx[sub][c] = f2a[fg * MAXFG + c];
      if (c == MAXFG) sMap[sub] = mapping[fg];
    }
    __syncthreads();
    if (fg < NFG && c < HSUB) {
      float s = 0.0f;
      #pragma unroll
      for (int j = 0; j < MAXFG; ++j) {
        const int row = sIdx[sub][j];
        if (row > 0)
          s += inAtom[(size_t)(row - 1) * NDIM + colBase + c];
      }
      atomicAdd(&acc[sMap[sub] * HSUB + c], s);   // ds_add_f32
    }
    __syncthreads();
  }

  // Flush per-block partials with hardware float atomics.
  for (int i = tid; i < NFUNC * HSUB; i += 256) {
    const float v = acc[i];
    if (v != 0.0f) {
      const int f = i / HSUB, cc = i % HSUB;
      unsafeAtomicAdd(&outp[f * NDIM + colBase + cc], v);  // global_atomic_add_f32
    }
  }
}

// ---------------------------------------------------------------------------
// Kernel 0: d_out = func_save_init (harness poisons d_out; atomics add on top)
// ---------------------------------------------------------------------------
__global__ void init_out(const float* __restrict__ ini, float* __restrict__ outp, int n) {
  const int i = blockIdx.x * blockDim.x + threadIdx.x;
  if (i < n) outp[i] = ini[i];
}

extern "C" void kernel_launch(void* const* d_in, const int* in_sizes, int n_in,
                              void* d_out, int out_size, void* d_ws, size_t ws_size,
                              hipStream_t stream) {
  const float* f_atoms   = (const float*)d_in[0];
  const float* W         = (const float*)d_in[1];
  const int*   func2atom = (const int*)d_in[2];
  const int*   mapping   = (const int*)d_in[3];
  const float* func_init = (const float*)d_in[4];
  float* outp = (float*)d_out;

  // ws layout: [0, 100KB) W fragment buffer; [100KB, ...) input_atom fp32.
  unsigned int* wfrag  = (unsigned int*)d_ws;
  float*        inAtom = (float*)((char*)d_ws + WFRAG_UINTS * sizeof(unsigned int));

  init_out<<<(out_size + 255) / 256, 256, 0, stream>>>(func_init, outp, out_size);
  wprep<<<(WFRAG_UINTS + 255) / 256, 256, 0, stream>>>(W, wfrag);
  gemm_f16_wmma<<<N_ATOMS / BM, 256, 0, stream>>>(f_atoms, wfrag, inAtom);
  gather_segsum<<<dim3(128, 3), 256, 0, stream>>>(inAtom, func2atom, mapping, outp);
}
